// UserModel_44401371906508
// MI455X (gfx1250) — matrix operand order repaired and checked by reading, not compiled, not run on hardware
//
#include <hip/hip_runtime.h>
#include <hip/hip_bf16.h>

// ---------------------------------------------------------------------------
// Sizes (fixed by the reference)
// ---------------------------------------------------------------------------
#define B_   64
#define T_   200
#define V_   128
#define K2_  2000   // NUM_C2
#define ND_  1000   // NUM_D

typedef __attribute__((ext_vector_type(16))) _Float16 v16h;
typedef __attribute__((ext_vector_type(8)))  float    v8f;

// Union view: fill a v16h WMMA operand via packed 32-bit LDS loads or
// element-wise f32->f16 conversion.
union AV { v16h v; unsigned int u[8]; _Float16 h[16]; };

static __device__ __forceinline__ void wait_async0() {
#if __has_builtin(__builtin_amdgcn_s_wait_asynccnt)
    __builtin_amdgcn_s_wait_asynccnt(0);
#else
    asm volatile("s_wait_asynccnt 0" ::: "memory");
#endif
}

// ---------------------------------------------------------------------------
// Kernel 1: fold the rank-1 structure.
//   x = [gamma*v_d, r*v_r]  =>  x @ W_ih.T = gamma*u_d + r*u_r
//   concat(beta*v_c, gamma*v_d, r*v_r) @ W2a.T = beta*w_c + gamma*w_d + r*w_r
// ws layout (floats): [0:384]=u_d [384:768]=u_r [768:896]=w_c [896:1024]=w_d
//                     [1024:1152]=w_r
// ---------------------------------------------------------------------------
__global__ void precompute_kernel(const float* __restrict__ W_ih,
                                  const float* __restrict__ W2a,
                                  const float* __restrict__ v_d,
                                  const float* __restrict__ v_r,
                                  const float* __restrict__ v_c,
                                  float* __restrict__ ws) {
    int tid = threadIdx.x;
    if (tid < 384) {
        float s1 = 0.f, s2 = 0.f;
        for (int k = 0; k < 128; ++k) {
            s1 += W_ih[tid * 256 + k]       * v_d[k];
            s2 += W_ih[tid * 256 + 128 + k] * v_r[k];
        }
        ws[tid]       = s1;
        ws[384 + tid] = s2;
    }
    if (tid < 128) {
        float sc = 0.f, sd = 0.f, sr = 0.f;
        for (int k = 0; k < 128; ++k) {
            sc += W2a[tid * 384 + k]       * v_c[k];
            sd += W2a[tid * 384 + 128 + k] * v_d[k];
            sr += W2a[tid * 384 + 256 + k] * v_r[k];
        }
        ws[768 + tid]  = sc;
        ws[896 + tid]  = sd;
        ws[1024 + tid] = sr;
    }
}

// ---------------------------------------------------------------------------
// Kernel 2: GRU recurrence. Single workgroup (8 wave32).
//   gh[64,384] = h[64,128] @ W_hh.T  via v_wmma_f32_16x16x32_f16, 200 steps.
// Work mapping: wave w owns hidden-column block j = 16w + ln for ALL 4 M-tiles
// and ALL 3 gates. Consequences:
//   * B operand (W_hh.T tiles) for this wave = 3 gates x 4 ktiles = 12 tile
//     sets = 96 VGPRs, loaded ONCE from global (f32->f16) and held in
//     registers for the whole 200-step loop: zero per-step B traffic.
//   * gh_r/gh_z/gh_n for a given (b,j) land in this lane's accumulators, so
//     the gate nonlinearity runs entirely in registers.
//   * gi constants (u_d/u_r/b_ih/b_hh at this lane's fixed j) hoisted too.
// Per-step LDS traffic is only the 16 A-tile loads of h_f16 + gate reads.
// ---------------------------------------------------------------------------
__global__ __launch_bounds__(256) void gru_kernel(
        const int*   __restrict__ d_seq,
        const float* __restrict__ r_seq,
        const float* __restrict__ D_emb,
        const float* __restrict__ W_hh,
        const float* __restrict__ b_ih,
        const float* __restrict__ b_hh,
        const float* __restrict__ ws,
        float*       __restrict__ out_h) {
    __shared__ _Float16 hf16[64 * 128];   // 16 KB : WMMA A operand
    __shared__ float    hf32[64 * 128];   // 32 KB : f32 recurrence state
    __shared__ float    demb[ND_];        //  4 KB : embedding table
    __shared__ float    gsh[64], rsh[64]; // per-batch gamma, r for this step

    int tid = threadIdx.x;
    for (int i = tid; i < 64 * 128; i += 256) { hf16[i] = (_Float16)0.f; hf32[i] = 0.f; }
    for (int i = tid; i < ND_; i += 256) demb[i] = D_emb[i];

    const int lane = tid & 31;
    const int w    = tid >> 5;
    const int half = lane >> 4;        // 0: lanes 0-15, 1: lanes 16-31
    const int ln   = lane & 15;
    const int jcol = w * 16 + ln;      // this lane's hidden column (0..127)

    // ---- Hoist B tiles for the whole t-loop (ISA B layout: lane -> N,
    //      VGPR v -> K = (v/4)*16 + half*8 + (v%4)*2; B[kk][j] = W_hh[j][kk]) -------------
    AV breg[3][4];
    #pragma unroll
    for (int g = 0; g < 3; ++g) {
        const int j = g * 128 + jcol;
        #pragma unroll
        for (int kt = 0; kt < 4; ++kt) {
            const int kb0 = kt * 32 + half * 8;
            #pragma unroll
            for (int v = 0; v < 16; ++v)
                breg[g][kt].h[v] = (_Float16)W_hh[j * 128 + kb0 + (v >> 3) * 16 + (v & 7)];
        }
    }
    // ---- Hoist per-lane gi constants ------------------------------------
    float udj[3], urj[3], bihj[3], bhhj[3];
    #pragma unroll
    for (int g = 0; g < 3; ++g) {
        udj[g]  = ws[g * 128 + jcol];
        urj[g]  = ws[384 + g * 128 + jcol];
        bihj[g] = b_ih[g * 128 + jcol];
        bhhj[g] = b_hh[g * 128 + jcol];
    }
    __syncthreads();

    const unsigned int* hf16u = (const unsigned int*)hf16;

    for (int t = 0; t < T_; ++t) {
        if (tid < 64) {
            gsh[tid] = demb[d_seq[tid * T_ + t]];
            rsh[tid] = r_seq[tid * T_ + t];
        }
        __syncthreads();   // gamma/r staged AND prev-step h_f16 committed

        v8f acc[4][3] = {};   // [mtile][gate r/z/n]

        #pragma unroll
        for (int mt = 0; mt < 4; ++mt) {
            const int m = mt * 16 + ln;
            #pragma unroll
            for (int kt = 0; kt < 4; ++kt) {
                // A operand: ISA 16-bit A layout:
                // VGPR v -> K = (v/4)*16 + half*8 + (v%4)*2
                AV a;
                #pragma unroll
                for (int v = 0; v < 8; ++v) {
                    int kb = kt * 32 + (v >> 2) * 16 + half * 8 + (v & 3) * 2;
                    a.u[v] = hf16u[(m * 128 + kb) >> 1];
                }
                #pragma unroll
                for (int g = 0; g < 3; ++g)
                    acc[mt][g] = __builtin_amdgcn_wmma_f32_16x16x32_f16(
                        false, a.v, false, breg[g][kt].v, (short)0,
                        acc[mt][g], false, false);
            }
        }
        __syncthreads();   // all A-reads of h_f16 done before we overwrite h

        // Gate math in registers. C/D layout: VGPR r -> M = r + 8*half, N = ln.
        #pragma unroll
        for (int mt = 0; mt < 4; ++mt) {
            #pragma unroll
            for (int r = 0; r < 8; ++r) {
                int brow = mt * 16 + half * 8 + r;
                float gamma = gsh[brow], rr = rsh[brow];
                float gir = gamma * udj[0] + rr * urj[0] + bihj[0];
                float giz = gamma * udj[1] + rr * urj[1] + bihj[1];
                float gin = gamma * udj[2] + rr * urj[2] + bihj[2];
                float ghr = acc[mt][0][r] + bhhj[0];
                float ghz = acc[mt][1][r] + bhhj[1];
                float ghn = acc[mt][2][r] + bhhj[2];
                float rg = 1.f / (1.f + __expf(-(gir + ghr)));
                float zg = 1.f / (1.f + __expf(-(giz + ghz)));
                float ng = tanhf(gin + rg * ghn);
                float hold = hf32[brow * 128 + jcol];
                float hnew = (1.f - zg) * ng + zg * hold;
                hf32[brow * 128 + jcol] = hnew;
                hf16[brow * 128 + jcol] = (_Float16)hnew;
                out_h[(brow * T_ + t) * V_ + jcol] = hnew;
            }
        }
        __syncthreads();   // protect gsh/rsh + h for next iteration
    }
}

// ---------------------------------------------------------------------------
// Kernel 3: alpha head. 200 blocks, each does a 64x128 row-tile of h_seq:
//   alpha = relu(h @ W1a.T + b1a) @ W1b.T + b1b
// WMMA GEMM with W1a f16 in LDS; fused ReLU * W1b reduction via ds_add_f32.
// h_seq rows are already flat [B*T][V] in output order.
// ---------------------------------------------------------------------------
__global__ __launch_bounds__(256) void alpha_kernel(
        const float* __restrict__ h_seq,
        const float* __restrict__ W1a,
        const float* __restrict__ b1a,
        const float* __restrict__ W1b,
        const float* __restrict__ b1b,
        float*       __restrict__ out_alpha) {
    __shared__ _Float16 w1[128 * 128];   // 32 KB [j][k]
    __shared__ _Float16 ht[64 * 128];    // 16 KB
    __shared__ float    b1s[128], w1bs[128], alpha[64];

    int tid  = threadIdx.x;
    int row0 = blockIdx.x * 64;

    for (int i = tid; i < 128 * 128; i += 256) w1[i] = (_Float16)W1a[i];
    for (int i = tid; i < 64 * 128; i += 256)
        ht[i] = (_Float16)h_seq[(size_t)row0 * 128 + i];
    if (tid < 128) { b1s[tid] = b1a[tid]; w1bs[tid] = W1b[tid]; }
    if (tid < 64)  alpha[tid] = 0.f;
    __syncthreads();

    const int lane  = tid & 31;
    const int w     = tid >> 5;
    const int half  = lane >> 4;
    const int ln    = lane & 15;
    const int mtile = w >> 1;
    const int nbase = (w & 1) * 4;
    const unsigned int* htu = (const unsigned int*)ht;
    const unsigned int* w1u = (const unsigned int*)w1;

    v8f acc[4] = {};
    const int m = mtile * 16 + ln;
    #pragma unroll
    for (int kt = 0; kt < 4; ++kt) {
        AV a;
        #pragma unroll
        for (int v = 0; v < 8; ++v) {
            int kb = kt * 32 + (v >> 2) * 16 + half * 8 + (v & 3) * 2;
            a.u[v] = htu[(m * 128 + kb) >> 1];
        }
        #pragma unroll
        for (int nt = 0; nt < 4; ++nt) {
            int j = (nbase + nt) * 16 + ln;
            AV b;
            int kb0 = kt * 32 + half * 8;
            #pragma unroll
            for (int v = 0; v < 8; ++v)
                b.u[v] = w1u[(j * 128 + kb0 + (v >> 2) * 16 + (v & 3) * 2) >> 1];
            acc[nt] = __builtin_amdgcn_wmma_f32_16x16x32_f16(
                false, a.v, false, b.v, (short)0, acc[nt], false, false);
        }
    }

    // Fused ReLU + dot with W1b, reduced across hidden dims via LDS atomics.
    #pragma unroll
    for (int r = 0; r < 8; ++r) {
        int mrow = mtile * 16 + half * 8 + r;
        float s = 0.f;
        #pragma unroll
        for (int nt = 0; nt < 4; ++nt) {
            int j = (nbase + nt) * 16 + ln;
            float v = acc[nt][r] + b1s[j];
            v = v > 0.f ? v : 0.f;
            s += v * w1bs[j];
        }
        atomicAdd(&alpha[mrow], s);
    }
    __syncthreads();
    if (tid < 64) out_alpha[row0 + tid] = alpha[tid] + b1b[0];
}

// ---------------------------------------------------------------------------
// Kernel 4: sequential memory chain + C_seq snapshots (the bandwidth roofline:
// 102 MB of stores). One block per batch chain.
// Per step: rank-1 hidden (beta*w_c + gamma*w_d + r*w_r + b2a), relu, dot W2b
// (computed redundantly per wave via shfl_xor -> no broadcast barrier), scatter
// into C_sh[c_t], then stream the 2000-float state LDS -> global with CDNA5
// async stores (global_store_async_from_lds_b128, ASYNCcnt-tracked) so the
// snapshot never round-trips through VGPRs.
// ---------------------------------------------------------------------------
__global__ __launch_bounds__(256) void mem_kernel(
        const int*   __restrict__ c_seq,
        const int*   __restrict__ d_seq,
        const float* __restrict__ r_seq,
        const float* __restrict__ D_emb,
        const float* __restrict__ ws,
        const float* __restrict__ b2a,
        const float* __restrict__ W2b,
        const float* __restrict__ b2b,
        float*       __restrict__ out_C) {
    __shared__ float C_sh[K2_];
    __shared__ float wc[128], wd[128], wr[128], b2s[128], w2bs[128];

    int tid  = threadIdx.x;
    int b    = blockIdx.x;
    int lane = tid & 31;

    for (int i = tid; i < K2_; i += 256) C_sh[i] = 0.f;
    if (tid < 128) {
        wc[tid]   = ws[768 + tid];
        wd[tid]   = ws[896 + tid];
        wr[tid]   = ws[1024 + tid];
        b2s[tid]  = b2a[tid];
        w2bs[tid] = W2b[tid];
    }
    __syncthreads();

    const float    bias2b  = b2b[0];
    // Low 32 bits of a flat shared address = LDS byte offset (aperture is in
    // the high bits), which is what the async-store VSRC operand wants.
    const unsigned ldsbase = (unsigned)(uintptr_t)&C_sh[0];
    const int i0 = tid, i1 = tid + 256;   // this thread's float4 slots (<500)

    for (int t = 0; t < T_; ++t) {
        int   c     = c_seq[b * T_ + t];
        float gamma = D_emb[d_seq[b * T_ + t]];
        float rr    = r_seq[b * T_ + t];
        float beta  = C_sh[c];

        // Redundant per-wave dot: lane handles j = lane + 32q.
        float p = 0.f;
        #pragma unroll
        for (int q = 0; q < 4; ++q) {
            int j = lane + 32 * q;
            float hv = beta * wc[j] + gamma * wd[j] + rr * wr[j] + b2s[j];
            hv = hv > 0.f ? hv : 0.f;
            p += hv * w2bs[j];
        }
        #pragma unroll
        for (int off = 16; off > 0; off >>= 1) p += __shfl_xor(p, off, 32);
        float new_c = p + bias2b;

        // Drain this wave's previous async snapshot (it reads LDS) before the
        // state is mutated; barrier makes every wave's drain global.
        wait_async0();
        __syncthreads();
        if (tid == 0) C_sh[c] = new_c;
        __syncthreads();

        // Async snapshot: 2000 floats = 500 x b128, straight from LDS.
        uint64_t gbase = (uint64_t)(uintptr_t)(out_C + (size_t)(b * T_ + t) * K2_);
        asm volatile("global_store_async_from_lds_b128 %0, %1, off"
                     :: "v"(gbase + 16ull * (unsigned)i0),
                        "v"(ldsbase + 16u * (unsigned)i0)
                     : "memory");
        if (i1 < K2_ / 4) {
            asm volatile("global_store_async_from_lds_b128 %0, %1, off"
                         :: "v"(gbase + 16ull * (unsigned)i1),
                            "v"(ldsbase + 16u * (unsigned)i1)
                         : "memory");
        }
    }
    // S_ENDPGM performs an implicit wait-idle, draining outstanding ASYNCcnt.
}

// ---------------------------------------------------------------------------
// Launch
// ---------------------------------------------------------------------------
extern "C" void kernel_launch(void* const* d_in, const int* in_sizes, int n_in,
                              void* d_out, int out_size, void* d_ws, size_t ws_size,
                              hipStream_t stream) {
    (void)in_sizes; (void)n_in; (void)out_size; (void)ws_size;

    const int*   c_seq = (const int*)  d_in[0];
    const int*   d_seq = (const int*)  d_in[1];
    const float* r_seq = (const float*)d_in[2];
    const float* D_emb = (const float*)d_in[3];
    const float* v_d   = (const float*)d_in[4];
    const float* v_r   = (const float*)d_in[5];
    const float* v_c   = (const float*)d_in[6];
    const float* W_ih  = (const float*)d_in[7];
    const float* W_hh  = (const float*)d_in[8];
    const float* b_ih  = (const float*)d_in[9];
    const float* b_hh  = (const float*)d_in[10];
    const float* W1a   = (const float*)d_in[11];
    const float* b1a   = (const float*)d_in[12];
    const float* W1b   = (const float*)d_in[13];
    const float* b1b   = (const float*)d_in[14];
    const float* W2a   = (const float*)d_in[15];
    const float* b2a   = (const float*)d_in[16];
    const float* W2b   = (const float*)d_in[17];
    const float* b2b   = (const float*)d_in[18];

    float* out       = (float*)d_out;
    float* out_alpha = out;                       // [B*T]        = 12800
    float* out_h     = out + 12800;               // [B*T, V]     = 1638400
    float* out_C     = out + 12800 + 1638400;     // [B*T, 2000]  = 25600000
    float* ws        = (float*)d_ws;

    precompute_kernel<<<1, 384, 0, stream>>>(W_ih, W2a, v_d, v_r, v_c, ws);
    gru_kernel<<<1, 256, 0, stream>>>(d_seq, r_seq, D_emb, W_hh, b_ih, b_hh, ws, out_h);
    alpha_kernel<<<200, 256, 0, stream>>>(out_h, W1a, b1a, W1b, b1b, out_alpha);
    mem_kernel<<<64, 256, 0, stream>>>(c_seq, d_seq, r_seq, D_emb, ws, b2a, W2b, b2b, out_C);
}
